// HungarianMatcher_635655159996
// MI455X (gfx1250) — matrix operand order repaired and checked by reading, not compiled, not run on hardware
//
#include <hip/hip_runtime.h>

typedef __attribute__((ext_vector_type(16))) _Float16 v16h;
typedef __attribute__((ext_vector_type(8)))  _Float16 v8h;
typedef __attribute__((ext_vector_type(8)))  float    v8f;
typedef __attribute__((ext_vector_type(8)))  unsigned int v8u;

#define BS   64
#define NQ   900
#define NT   300
#define NC   80
#define KPAD 96
#define EPS  1e-7f

__device__ __forceinline__ unsigned int lds_off(const void* p) {
    // LDS aperture lives in addr[63:32]; addr[31:0] is the offset within the
    // workgroup's LDS allocation (see CDNA5 flat aperture mapping).
    return (unsigned int)(unsigned long long)p;
}

__device__ __forceinline__ void async_ld_b128(unsigned int ldst, const void* g) {
    asm volatile("global_load_async_to_lds_b128 %0, %1, off"
                 :: "v"(ldst), "v"((unsigned long long)g) : "memory");
}

// ---------------------------------------------------------------------------
// Kernel 1: row softmax (f32 in, f16 out, K padded 80 -> 96 with zeros).
// One wave32 per row; lanes cover c = {lane, lane+32, lane+64}.
// ---------------------------------------------------------------------------
__global__ void hm_softmax_f16(const float* __restrict__ logits,
                               _Float16* __restrict__ p16) {
    const int row  = blockIdx.x * 8 + (threadIdx.x >> 5);
    const int lane = threadIdx.x & 31;
    if (row >= BS * NQ) return;

    const float* lrow = logits + (size_t)row * NC;
    float l0 = lrow[lane];
    float l1 = lrow[lane + 32];
    float l2 = (lane < 16) ? lrow[lane + 64] : -3.0e38f;   // c in [64,80)

    float m = fmaxf(fmaxf(l0, l1), l2);
    #pragma unroll
    for (int off = 16; off > 0; off >>= 1) m = fmaxf(m, __shfl_xor(m, off, 32));

    float e0 = __expf(l0 - m);
    float e1 = __expf(l1 - m);
    float e2 = (lane < 16) ? __expf(l2 - m) : 0.0f;
    float s  = e0 + e1 + e2;
    #pragma unroll
    for (int off = 16; off > 0; off >>= 1) s += __shfl_xor(s, off, 32);

    const float inv = __builtin_amdgcn_rcpf(s);
    _Float16* prow = p16 + (size_t)row * KPAD;
    prow[lane]      = (_Float16)(e0 * inv);
    prow[lane + 32] = (_Float16)(e1 * inv);
    prow[lane + 64] = (_Float16)((lane < 16) ? e2 * inv : 0.0f);  // zero pad 80..95
}

// ---------------------------------------------------------------------------
// Kernel 2: fused cost matrix.
// Block = 256 thr = 8 waves arranged 2 (q) x 4 (t): tile = 32q x 64t.
// Stage raw boxes/classes via global_load_async_to_lds_b128 (ASYNCcnt),
// one 16x16 WMMA tile per wave for the class term (K=96 -> 3 wmma f16),
// then VALU for L1 + GIoU with fast rcp.
// ---------------------------------------------------------------------------
__global__ void __launch_bounds__(256)
hm_cost_kernel(const float* __restrict__ pred_boxes,
               const float* __restrict__ tgt_boxes,
               const int*   __restrict__ tgt_classes,
               const float* __restrict__ img_whwh,
               const _Float16* __restrict__ p16,
               float* __restrict__ out) {
    const int b      = blockIdx.z;
    const int q_base = blockIdx.x * 32;
    const int t_base = blockIdx.y * 64;
    const int tid    = threadIdx.x;
    const int lane   = tid & 31;
    const int wave   = tid >> 5;

    __shared__ __align__(16) float sPredB[32][4];   // absolute xyxy
    __shared__ __align__(16) float sTgtB[64][4];
    __shared__ __align__(16) int   sTgtC[64];
    __shared__ __align__(16) float sWhwh[4];
    __shared__ float sPredA[32];                    // absolute areas
    __shared__ float sTgtA[64];

    // ---- Phase 1: async-stage raw tiles into LDS (addresses edge-clamped) --
    if (tid < 32) {
        int q = q_base + tid; if (q > NQ - 1) q = NQ - 1;
        async_ld_b128(lds_off(&sPredB[tid][0]),
                      pred_boxes + ((size_t)b * NQ + q) * 4);
    } else if (tid < 96) {
        int ti = tid - 32;
        int t = t_base + ti; if (t > NT - 1) t = NT - 1;
        async_ld_b128(lds_off(&sTgtB[ti][0]),
                      tgt_boxes + ((size_t)b * NT + t) * 4);
    } else if (tid < 112) {
        int li = tid - 96;                           // 4 classes per lane
        int t0 = t_base + 4 * li; if (t0 > NT - 4) t0 = NT - 4;
        async_ld_b128(lds_off(&sTgtC[4 * li]),
                      tgt_classes + (size_t)b * NT + t0);
    } else if (tid == 112) {
        async_ld_b128(lds_off(&sWhwh[0]), img_whwh + (size_t)b * 4);
    }
    asm volatile("s_wait_asynccnt 0x0" ::: "memory");
    __syncthreads();

    // ---- Phase 2: derive areas in LDS ----
    if (tid < 32) {
        sPredA[tid] = (sPredB[tid][2] - sPredB[tid][0]) *
                      (sPredB[tid][3] - sPredB[tid][1]);
    } else if (tid < 96) {
        int ti = tid - 32;
        sTgtA[ti] = (sTgtB[ti][2] - sTgtB[ti][0]) *
                    (sTgtB[ti][3] - sTgtB[ti][1]);
    }
    __syncthreads();

    const float invW = __builtin_amdgcn_rcpf(sWhwh[0]);
    const float invH = __builtin_amdgcn_rcpf(sWhwh[1]);

    const int waveQ = wave >> 2;          // 0..1
    const int waveT = wave & 3;           // 0..3
    const int h     = lane >> 4;          // lane half
    const int ln    = lane & 15;

    // ---- A fragments: 16 rows of P16, K chunks {0,32,64}. Layout:
    // lane half h, element e <-> K = k0+16h+e
    int qrow = q_base + waveQ * 16 + ln; if (qrow > NQ - 1) qrow = NQ - 1;
    const _Float16* arow = p16 + ((size_t)b * NQ + qrow) * KPAD;

    v8h lo, hi;
    lo = *(const v8h*)(arow + 0  + 16 * h);
    hi = *(const v8h*)(arow + 8 + 16 * h);
    v16h a0 = __builtin_shufflevector(lo, hi, 0,1,2,3,4,5,6,7,8,9,10,11,12,13,14,15);
    lo = *(const v8h*)(arow + 32 + 16 * h);
    hi = *(const v8h*)(arow + 40 + 16 * h);
    v16h a1 = __builtin_shufflevector(lo, hi, 0,1,2,3,4,5,6,7,8,9,10,11,12,13,14,15);
    lo = *(const v8h*)(arow + 64 + 16 * h);
    hi = *(const v8h*)(arow + 72 + 16 * h);
    v16h a2 = __builtin_shufflevector(lo, hi, 0,1,2,3,4,5,6,7,8,9,10,11,12,13,14,15);

    // ---- B fragments: one-hot(class). Element e <-> K = k0 + 16h + e, so
    // VGPR j holds K-pair (k0+16h+2j, +1). Fold the lane half into the class
    // pair index once: d = (c>>1) - 8h, then every compare is against an
    // inline constant (j / 16+j / 32+j) -> v_cmp + v_cndmask per VGPR.
    const int tl = waveT * 16 + ln;        // index within 64-entry t tile
    const int c  = sTgtC[tl];
    const unsigned int pat = (c & 1) ? 0x3C000000u : 0x00003C00u;
    const int d = (c >> 1) - 8 * h;
    v8u B0, B1, B2;
    #pragma unroll
    for (int j = 0; j < 8; ++j) {
        B0[j] = (d == j)      ? pat : 0u;   // K pairs  0..15
        B1[j] = (d == 16 + j) ? pat : 0u;   // K pairs 16..31
        B2[j] = (d == 32 + j) ? pat : 0u;   // K pairs 32..47 (>=40 never hits)
    }
    v16h b0 = __builtin_bit_cast(v16h, B0);
    v16h b1 = __builtin_bit_cast(v16h, B1);
    v16h b2 = __builtin_bit_cast(v16h, B2);

    // ---- class term: acc[m,n] = prob[q0+m, cls[t0+n]] ----
    v8f acc = {0.f, 0.f, 0.f, 0.f, 0.f, 0.f, 0.f, 0.f};
    acc = __builtin_amdgcn_wmma_f32_16x16x32_f16(false, a0, false, b0, (short)0, acc, false, false);
    acc = __builtin_amdgcn_wmma_f32_16x16x32_f16(false, a1, false, b1, (short)0, acc, false, false);
    acc = __builtin_amdgcn_wmma_f32_16x16x32_f16(false, a2, false, b2, (short)0, acc, false, false);

    // ---- L1 + GIoU on the 8 accumulator elements (M = r + 8h, N = ln) ----
    const float tx1 = sTgtB[tl][0], ty1 = sTgtB[tl][1];
    const float tx2 = sTgtB[tl][2], ty2 = sTgtB[tl][3];
    const float ta  = sTgtA[tl];
    const int   t_g = t_base + tl;
    const bool  tvalid = (t_g < NT);

    const int q0l  = q_base + waveQ * 16 + 8 * h;    // first q this lane touches
    const int qrem = NQ - q0l;                       // store iff r < qrem
    float* orow = out + ((size_t)b * NQ + q0l) * NT + t_g;

    #pragma unroll
    for (int r = 0; r < 8; ++r) {
        const int qi = waveQ * 16 + 8 * h + r;       // [0,32)

        const float px1 = sPredB[qi][0], py1 = sPredB[qi][1];
        const float px2 = sPredB[qi][2], py2 = sPredB[qi][3];
        const float pa  = sPredA[qi];

        // L1 on normalized boxes == scaled absolute differences
        float sx = fabsf(px1 - tx1) + fabsf(px2 - tx2);
        float sy = fabsf(py1 - ty1) + fabsf(py2 - ty2);
        float l1c = fmaf(sx, invW, sy * invH);

        float ix1 = fmaxf(px1, tx1), iy1 = fmaxf(py1, ty1);
        float ix2 = fminf(px2, tx2), iy2 = fminf(py2, ty2);
        float iw  = fmaxf(ix2 - ix1, 0.f), ih = fmaxf(iy2 - iy1, 0.f);
        float inter = iw * ih;
        float uni   = pa + ta - inter;
        float iour  = inter * __builtin_amdgcn_rcpf(uni + EPS);
        float iou   = (inter > 0.f) ? iour : 0.f;

        float cx1 = fminf(px1, tx1), cy1 = fminf(py1, ty1);
        float cx2 = fmaxf(px2, tx2), cy2 = fmaxf(py2, ty2);
        float cw  = fmaxf(cx2 - cx1, 0.f), ch = fmaxf(cy2 - cy1, 0.f);
        float ca  = cw * ch;
        float giou = iou - (ca - uni) * __builtin_amdgcn_rcpf(ca + EPS);

        float cost = fmaf(5.0f, l1c, -acc[r]) - 2.0f * giou;

        if (tvalid && r < qrem)
            orow[(size_t)r * NT] = cost;
    }
}

// ---------------------------------------------------------------------------
extern "C" void kernel_launch(void* const* d_in, const int* in_sizes, int n_in,
                              void* d_out, int out_size, void* d_ws, size_t ws_size,
                              hipStream_t stream) {
    (void)in_sizes; (void)n_in; (void)out_size; (void)ws_size;
    const float* pred_logits = (const float*)d_in[0];
    const float* pred_boxes  = (const float*)d_in[1];
    const float* tgt_boxes   = (const float*)d_in[2];
    const int*   tgt_classes = (const int*)d_in[3];
    const float* img_whwh    = (const float*)d_in[4];
    float* out = (float*)d_out;

    _Float16* p16 = (_Float16*)d_ws;   // BS*NQ*96 f16 = 11.06 MB scratch

    const int rows = BS * NQ;                       // 57600
    hm_softmax_f16<<<dim3((rows + 7) / 8), dim3(256), 0, stream>>>(pred_logits, p16);

    dim3 grid((NQ + 31) / 32, (NT + 63) / 64, BS);  // (29, 5, 64)
    hm_cost_kernel<<<grid, dim3(256), 0, stream>>>(pred_boxes, tgt_boxes,
                                                   tgt_classes, img_whwh, p16, out);
}
